// StructureModule_78640851189929
// MI455X (gfx1250) — matrix operand order repaired and checked by reading, not compiled, not run on hardware
//
#include <hip/hip_runtime.h>
#include <cstdint>
#include <cstddef>

// ---------------------------------------------------------------------------
// StructureModule for MI455X (gfx1250, wave32, WMMA).
// All heavy GEMMs use V_WMMA_F32_16X16X32_F16 (f16 in, f32 accumulate).
// z is LayerNormed once and cached as f16 (67MB -> resident in 192MB L2).
// Distance logits folded into the QK WMMA via the |q-k|^2 expansion trick.
// Out-of-range tile rows/cols are handled by pointer clamping (garbage lands
// only in never-stored accumulator slots) -> no divergence in WMMA loops.
// ---------------------------------------------------------------------------

typedef _Float16 h16;
typedef __attribute__((ext_vector_type(8)))  _Float16 v8h;
typedef __attribute__((ext_vector_type(16))) _Float16 v16h;
typedef __attribute__((ext_vector_type(8)))  float    v8f;

// ----------------------------- device helpers ------------------------------

static __device__ __forceinline__ v8f zero8() {
  v8f z;
#pragma unroll
  for (int e = 0; e < 8; ++e) z[e] = 0.f;
  return z;
}
// A-frag (16x32 f16, row-major source). Caller passes p = rowBase + k0 + (lane&16 ? 8 : 0).
// lane<16: elems 0..7 = K[0..7], 8..15 = K[16..23]; lane>=16: K[8..15], K[24..31].
static __device__ __forceinline__ v16h mk_a(const h16* p) {
  v8h lo = *(const v8h*)(p);
  v8h hi = *(const v8h*)(p + 16);
  v16h r;
#pragma unroll
  for (int e = 0; e < 8; ++e) { r[e] = lo[e]; r[e + 8] = hi[e]; }
  return r;
}
// B-frag (32x16 f16) from transposed weights Bt[n][k]. p = Bt + n*ldb + k0 + (lane&16 ? 8 : 0).
static __device__ __forceinline__ v16h mk_b(const h16* p) {
  v8h lo = *(const v8h*)(p);
  v8h hi = *(const v8h*)(p + 16);
  v16h r;
#pragma unroll
  for (int e = 0; e < 8; ++e) { r[e] = lo[e]; r[e + 8] = hi[e]; }
  return r;
}
static __device__ __forceinline__ v8f wmma16(v16h a, v16h b, v8f c) {
  return __builtin_amdgcn_wmma_f32_16x16x32_f16(false, a, false, b, (short)0, c, false, false);
}
static __device__ __forceinline__ float wred(float v) {
#pragma unroll
  for (int o = 16; o > 0; o >>= 1) v += __shfl_xor(v, o, 32);
  return v;
}
static __device__ __forceinline__ int imin(int a, int b) { return a < b ? a : b; }

// ----------------------------- kernels -------------------------------------

// f32 [din][dout] -> f16 transposed [dout][din]
__global__ void wtr_k(const float* __restrict__ W, h16* __restrict__ Wt, int din, int dout) {
  int idx = blockIdx.x * 256 + threadIdx.x;
  if (idx < din * dout) {
    int i = idx / dout, o = idx % dout;
    Wt[(size_t)o * din + i] = (h16)W[idx];
  }
}

// Generic tiled GEMM: C = A(MxK,f16) * Bt(NxK,f16)^T + bias (+Cin).
// One wave computes a 16x32 macro-tile: one A-frag feeds two WMMAs per k-step.
__global__ void gemm16_k(const h16* __restrict__ A, int lda,
                         const h16* __restrict__ Bt, int ldb,
                         const float* __restrict__ bias,
                         const float* __restrict__ Cin, int ldcin,
                         float* __restrict__ Cf, h16* __restrict__ Ch, int ldc,
                         int M, int K, int N, int relu) {
  const int lane = threadIdx.x;
  const int ntn  = (N + 31) >> 5;                 // macro columns (32 wide)
  const int m0   = (blockIdx.x / ntn) * 16;
  const int n0   = (blockIdx.x % ntn) * 32;
  const int nl0  = n0 + (lane & 15);
  const int nl1  = n0 + 16 + (lane & 15);
  const int moff = (lane & 16) ? 8 : 0;
  const int koffA = (lane & 16) ? 8 : 0;
  const int koffB = (lane & 16) ? 8 : 0;
  const h16* arow  = A + (size_t)(m0 + (lane & 15)) * lda + koffA;
  // Clamp out-of-range columns to a valid row: garbage only reaches
  // accumulator slots that are never stored (store is guarded by n < N).
  const h16* brow0 = Bt + (size_t)imin(nl0, N - 1) * ldb + koffB;
  const h16* brow1 = Bt + (size_t)imin(nl1, N - 1) * ldb + koffB;
  v8f acc0 = zero8();
  v8f acc1 = zero8();
  for (int k0 = 0; k0 < K; k0 += 32) {
    v16h a  = mk_a(arow + k0);
    v16h b0 = mk_b(brow0 + k0);
    v16h b1 = mk_b(brow1 + k0);
    acc0 = wmma16(a, b0, acc0);
    acc1 = wmma16(a, b1, acc1);
  }
  float bv0 = (bias && nl0 < N) ? bias[nl0] : 0.f;
  float bv1 = (bias && nl1 < N) ? bias[nl1] : 0.f;
#pragma unroll
  for (int r = 0; r < 8; ++r) {
    int m = m0 + r + moff;
    if (m < M) {
      if (nl0 < N) {
        float v = acc0[r] + bv0;
        if (Cin)  v += Cin[(size_t)m * ldcin + nl0];
        if (relu) v = fmaxf(v, 0.f);
        if (Cf) Cf[(size_t)m * ldc + nl0] = v;
        if (Ch) Ch[(size_t)m * ldc + nl0] = (h16)v;
      }
      if (nl1 < N) {
        float v = acc1[r] + bv1;
        if (Cin)  v += Cin[(size_t)m * ldcin + nl1];
        if (relu) v = fmaxf(v, 0.f);
        if (Cf) Cf[(size_t)m * ldc + nl1] = v;
        if (Ch) Ch[(size_t)m * ldc + nl1] = (h16)v;
      }
    }
  }
}

// Row LayerNorm (one wave per row), optional f32 and f16 outputs.
__global__ void ln_row_k(const float* __restrict__ x, const float* __restrict__ g,
                         const float* __restrict__ b, float* __restrict__ y32,
                         h16* __restrict__ y16, int C) {
  int r = blockIdx.x, lane = threadIdx.x;
  const float* xr = x + (size_t)r * C;
  float s = 0.f;
  for (int c = lane; c < C; c += 32) s += xr[c];
  float mean = wred(s) / (float)C;
  float v = 0.f;
  for (int c = lane; c < C; c += 32) { float d = xr[c] - mean; v += d * d; }
  float inv = rsqrtf(wred(v) / (float)C + 1e-5f);
  for (int c = lane; c < C; c += 32) {
    float yv = (xr[c] - mean) * inv * g[c] + b[c];
    if (y32) y32[(size_t)r * C + c] = yv;
    if (y16) y16[(size_t)r * C + c] = (h16)yv;
  }
}

// LN over z rows (128 ch), fused: write zl transposed f16 [i][c][j], and B = zl@Wb+bb as f16 [h][i][j].
__global__ void ln_z_k(const float* __restrict__ z, const float* __restrict__ g,
                       const float* __restrict__ b, const float* __restrict__ Wb,
                       const float* __restrict__ bbv, h16* __restrict__ zlT,
                       h16* __restrict__ Bterm) {
  int blk = blockIdx.x;
  int i = blk >> 9, j = blk & 511;
  int lane = threadIdx.x;
  const float* row = z + (((size_t)i * 512) + j) * 128;
  float zr[4]; float s = 0.f;
#pragma unroll
  for (int k = 0; k < 4; ++k) { zr[k] = row[lane + 32 * k]; s += zr[k]; }
  float mean = wred(s) * (1.f / 128.f);
  float var = 0.f;
#pragma unroll
  for (int k = 0; k < 4; ++k) { float d = zr[k] - mean; var += d * d; }
  float inv = rsqrtf(wred(var) * (1.f / 128.f) + 1e-5f);
  float zl[4];
#pragma unroll
  for (int k = 0; k < 4; ++k) {
    int c = lane + 32 * k;
    zl[k] = (zr[k] - mean) * inv * g[c] + b[c];
    zlT[((size_t)i * 128 + c) * 512 + j] = (h16)zl[k];
  }
  float bh[12];
#pragma unroll
  for (int h = 0; h < 12; ++h) {
    float p = 0.f;
#pragma unroll
    for (int k = 0; k < 4; ++k) p += zl[k] * Wb[(lane + 32 * k) * 12 + h];
    bh[h] = wred(p);
  }
  if (lane == 0) {
#pragma unroll
    for (int h = 0; h < 12; ++h)
      Bterm[((size_t)h * 512 + i) * 512 + j] = (h16)(bh[h] + bbv[h]);
  }
}

__global__ void frames_init_k(float* __restrict__ rots, float* __restrict__ trans) {
  int i = blockIdx.x * 64 + threadIdx.x;
  if (i < 512) {
#pragma unroll
    for (int e = 0; e < 9; ++e) rots[i * 9 + e] = (e == 0 || e == 4 || e == 8) ? 1.f : 0.f;
    trans[i * 3 + 0] = 0.f; trans[i * 3 + 1] = 0.f; trans[i * 3 + 2] = 0.f;
  }
}

// Per-residue attention operand prep: AttA=[ns*Q | hw*Q3rot | 0], AttB=[K | K3rot | 0],
// rowq/rowk = -0.5*hw*sum|pt|^2, VV3t=[V ; V3rot] transposed f16 [h][40][j].
__global__ void qkv_prep_k(const float* __restrict__ qraw, const float* __restrict__ kvraw,
                           const float* __restrict__ q3raw, const float* __restrict__ kv3raw,
                           const float* __restrict__ hwp, const float* __restrict__ rots,
                           const float* __restrict__ trans, h16* __restrict__ AttA,
                           h16* __restrict__ AttB, float* __restrict__ rowq,
                           float* __restrict__ rowk, h16* __restrict__ VV3t) {
  const int r = blockIdx.x;
  const int t = threadIdx.x;
  __shared__ float aq[12], ak[12];
  if (t < 12) { aq[t] = 0.f; ak[t] = 0.f; }
  __syncthreads();
  float R[9], T[3];
#pragma unroll
  for (int e = 0; e < 9; ++e) R[e] = rots[r * 9 + e];
#pragma unroll
  for (int e = 0; e < 3; ++e) T[e] = trans[r * 3 + e];
  const float nhw = rsqrtf(54.f);  // (3*NQK*9/2)^-0.5
  const float ns  = rsqrtf(48.f);  // (3*C_IPA)^-0.5
  if (t < 48) {                     // Q3 points: k=(h*4+p)
    int h = t >> 2, p = t & 3;
    float v0 = q3raw[r * 144 + t], v1 = q3raw[r * 144 + 48 + t], v2 = q3raw[r * 144 + 96 + t];
    float w0 = R[0] * v0 + R[1] * v1 + R[2] * v2 + T[0];
    float w1 = R[3] * v0 + R[4] * v1 + R[5] * v2 + T[1];
    float w2 = R[6] * v0 + R[7] * v1 + R[8] * v2 + T[2];
    float hws = log1pf(__expf(hwp[h])) * nhw;
    h16* dst = AttA + ((size_t)r * 12 + h) * 32 + 16 + p * 3;
    dst[0] = (h16)(hws * w0); dst[1] = (h16)(hws * w1); dst[2] = (h16)(hws * w2);
    atomicAdd(&aq[h], w0 * w0 + w1 * w1 + w2 * w2);
  } else if (t < 192) {             // KV3 points: k=(h*12+tt), tt<4 -> K3, else V3
    int k = t - 48; int h = k / 12; int tt = k % 12;
    float v0 = kv3raw[r * 432 + k], v1 = kv3raw[r * 432 + 144 + k], v2 = kv3raw[r * 432 + 288 + k];
    float w0 = R[0] * v0 + R[1] * v1 + R[2] * v2 + T[0];
    float w1 = R[3] * v0 + R[4] * v1 + R[5] * v2 + T[1];
    float w2 = R[6] * v0 + R[7] * v1 + R[8] * v2 + T[2];
    if (tt < 4) {
      h16* dst = AttB + ((size_t)r * 12 + h) * 32 + 16 + tt * 3;
      dst[0] = (h16)w0; dst[1] = (h16)w1; dst[2] = (h16)w2;
      atomicAdd(&ak[h], w0 * w0 + w1 * w1 + w2 * w2);
    } else {
      int c = tt - 4;
      VV3t[((size_t)h * 40 + 16 + c * 3 + 0) * 512 + r] = (h16)w0;
      VV3t[((size_t)h * 40 + 16 + c * 3 + 1) * 512 + r] = (h16)w1;
      VV3t[((size_t)h * 40 + 16 + c * 3 + 2) * 512 + r] = (h16)w2;
    }
  } else if (t < 384) {             // scalar Q/K/V channels
    int u = t - 192; int h = u >> 4; int c = u & 15;
    AttA[((size_t)r * 12 + h) * 32 + c] = (h16)(ns * qraw[r * 192 + u]);
    AttB[((size_t)r * 12 + h) * 32 + c] = (h16)(kvraw[r * 384 + h * 32 + c]);
    VV3t[((size_t)h * 40 + c) * 512 + r] = (h16)(kvraw[r * 384 + h * 32 + 16 + c]);
  } else if (t < 432) {             // zero pad K-dim 28..31
    int u = t - 384; int h = u >> 2; int e = u & 3;
    AttA[((size_t)r * 12 + h) * 32 + 28 + e] = (h16)0.f;
    AttB[((size_t)r * 12 + h) * 32 + 28 + e] = (h16)0.f;
  }
  __syncthreads();
  if (t < 12) {
    float hws = log1pf(__expf(hwp[t])) * nhw;
    rowq[r * 12 + t] = -0.5f * hws * aq[t];
    rowk[r * 12 + t] = -0.5f * hws * ak[t];
  }
}

// Attention logits via single WMMA per 16x16 tile (K=32 ext dim), softmax over j, P f16 out.
__global__ void attn_k(const h16* __restrict__ AttA, const h16* __restrict__ AttB,
                       const float* __restrict__ rowq, const float* __restrict__ rowk,
                       const h16* __restrict__ Bterm, const float* __restrict__ mask,
                       h16* __restrict__ P) {
  __shared__ float lg[16 * 512];
  const int i0 = blockIdx.x * 16;
  const int h  = blockIdx.y;
  const int lane = threadIdx.x;
  const float nb = 0.57735026919f;  // 3^-0.5
  const int moff = (lane & 16) ? 8 : 0;
  const h16* ap = AttA + ((size_t)(i0 + (lane & 15)) * 12 + h) * 32 + ((lane & 16) ? 8 : 0);
  v16h a = mk_a(ap);
  float rq[8], mi[8];
#pragma unroll
  for (int r = 0; r < 8; ++r) {
    int m = i0 + r + moff;
    rq[r] = rowq[m * 12 + h];
    mi[r] = mask[m];
  }
  for (int jt = 0; jt < 32; ++jt) {
    int n = jt * 16 + (lane & 15);
    const h16* bp = AttB + ((size_t)n * 12 + h) * 32 + ((lane & 16) ? 8 : 0);
    v8f c = wmma16(a, mk_b(bp), zero8());
    float rk = rowk[n * 12 + h];
    float mj = mask[n];
#pragma unroll
    for (int r = 0; r < 8; ++r) {
      int m = r + moff;
      float bt = (float)Bterm[((size_t)h * 512 + (i0 + m)) * 512 + n];
      lg[m * 512 + n] = c[r] + rq[r] + rk + nb * bt + 1e9f * (mi[r] * mj - 1.f);
    }
  }
  __syncthreads();
  int row = lane & 15;
  int cb  = (lane >> 4) * 256;
  float mx = -3.0e38f;
  for (int c = 0; c < 256; ++c) mx = fmaxf(mx, lg[row * 512 + cb + c]);
  mx = fmaxf(mx, __shfl_xor(mx, 16, 32));
  float sum = 0.f;
  for (int c = 0; c < 256; ++c) sum += __expf(lg[row * 512 + cb + c] - mx);
  sum += __shfl_xor(sum, 16, 32);
  float inv = 1.f / sum;
  h16* pr = P + (size_t)(i0 + row) * 6144 + (size_t)h * 512 + cb;
  for (int c = 0; c < 256; ++c) pr[c] = (h16)(__expf(lg[row * 512 + cb + c] - mx) * inv);
}

// O (N=16) and O3 (N=24) : [16 x 40] = P[itile,h,:] @ VV3t[h]^T over K=512.
__global__ void oo3_k(const h16* __restrict__ P, const h16* __restrict__ VV3t,
                      h16* __restrict__ Of, float* __restrict__ O3raw) {
  const int i0 = blockIdx.x * 16;
  const int h  = blockIdx.y;
  const int lane = threadIdx.x;
  const int moff = (lane & 16) ? 8 : 0;
  const h16* arow = P + (size_t)(i0 + (lane & 15)) * 6144 + (size_t)h * 512 + ((lane & 16) ? 8 : 0);
  const int koffB = (lane & 16) ? 8 : 0;
#pragma unroll
  for (int nt = 0; nt < 3; ++nt) {
    int n = nt * 16 + (lane & 15);
    const h16* brow = VV3t + ((size_t)h * 40 + imin(n, 39)) * 512 + koffB;  // clamp cols 40..47
    v8f acc = zero8();
    for (int k0 = 0; k0 < 512; k0 += 32) {
      acc = wmma16(mk_a(arow + k0), mk_b(brow + k0), acc);
    }
#pragma unroll
    for (int rr = 0; rr < 8; ++rr) {
      int m = i0 + rr + moff;
      if (n < 16)      Of[(size_t)m * 2112 + h * 16 + n]           = (h16)acc[rr];
      else if (n < 40) O3raw[(size_t)m * 288 + h * 24 + (n - 16)]  = acc[rr];
    }
  }
}

// Op[i,h,c] = sum_j P[i,h,j] * zl[i,j,c] : per-residue M=12(pad16), K=512, N=16 per tile.
__global__ void op_k(const h16* __restrict__ P, const h16* __restrict__ zlT, h16* __restrict__ Of) {
  const int i = blockIdx.x;
  const int ct = blockIdx.y;
  const int lane = threadIdx.x;
  const int hr = imin(lane & 15, 11);  // clamp pad rows 12..15 (never stored)
  const int moff = (lane & 16) ? 8 : 0;
  const h16* arow = P + (size_t)i * 6144 + (size_t)hr * 512 + ((lane & 16) ? 8 : 0);
  const h16* brow = zlT + ((size_t)i * 128 + ct * 16 + (lane & 15)) * 512 + ((lane & 16) ? 8 : 0);
  v8f acc = zero8();
  for (int k0 = 0; k0 < 512; k0 += 32) {
    acc = wmma16(mk_a(arow + k0), mk_b(brow + k0), acc);
  }
#pragma unroll
  for (int rr = 0; rr < 8; ++rr) {
    int h = rr + moff;
    if (h < 12)
      Of[(size_t)i * 2112 + 576 + (size_t)h * 128 + ct * 16 + (lane & 15)] = (h16)acc[rr];
  }
}

// Inverse-frame transform of O3, norms, scatter into Of concat slots.
__global__ void o3fin_k(const float* __restrict__ O3raw, const float* __restrict__ rots,
                        const float* __restrict__ trans, h16* __restrict__ Of) {
  const int i = blockIdx.x;
  const int g = threadIdx.x;  // 96 = 12 heads * 8 value points
  const float* R = rots + i * 9;
  const float* T = trans + i * 3;
  const float* p = O3raw + (size_t)i * 288 + (g >> 3) * 24 + (g & 7) * 3;
  float d0 = p[0] - T[0], d1 = p[1] - T[1], d2 = p[2] - T[2];
  float x = R[0] * d0 + R[3] * d1 + R[6] * d2;
  float y = R[1] * d0 + R[4] * d1 + R[7] * d2;
  float z = R[2] * d0 + R[5] * d1 + R[8] * d2;
  float n = sqrtf(x * x + y * y + z * z + 1e-20f);
  h16* o = Of + (size_t)i * 2112 + 192;
  o[g] = (h16)x; o[96 + g] = (h16)y; o[192 + g] = (h16)z; o[288 + g] = (h16)n;
}

// Backbone update: BB = s@Wbb+bbb, quaternion rot, compose frames.
__global__ void bbq_k(const float* __restrict__ s, const float* __restrict__ Wbb,
                      const float* __restrict__ bbb, float* __restrict__ rots,
                      float* __restrict__ trans) {
  const int r = blockIdx.x;
  const int lane = threadIdx.x;
  float p[6];
#pragma unroll
  for (int o = 0; o < 6; ++o) {
    float acc = 0.f;
    for (int c = lane; c < 384; c += 32) acc += s[(size_t)r * 384 + c] * Wbb[c * 6 + o];
    p[o] = wred(acc);
  }
  if (lane == 0) {
    float b = p[0] + bbb[0], c = p[1] + bbb[1], d = p[2] + bbb[2];
    float t0 = p[3] + bbb[3], t1 = p[4] + bbb[4], t2 = p[5] + bbb[5];
    float a = 1.f;
    float n = rsqrtf(a * a + b * b + c * c + d * d);
    a *= n; b *= n; c *= n; d *= n;
    float Ru[9] = { a*a + b*b - c*c - d*d, 2.f*(b*c - a*d),       2.f*(b*d + a*c),
                    2.f*(b*c + a*d),       a*a - b*b + c*c - d*d, 2.f*(c*d - a*b),
                    2.f*(b*d - a*c),       2.f*(c*d + a*b),       a*a - b*b - c*c + d*d };
    float R[9], T[3];
#pragma unroll
    for (int e = 0; e < 9; ++e) R[e] = rots[r * 9 + e];
#pragma unroll
    for (int e = 0; e < 3; ++e) T[e] = trans[r * 3 + e];
    float tn0 = R[0] * t0 + R[1] * t1 + R[2] * t2 + T[0];
    float tn1 = R[3] * t0 + R[4] * t1 + R[5] * t2 + T[1];
    float tn2 = R[6] * t0 + R[7] * t1 + R[8] * t2 + T[2];
    float Rn[9];
#pragma unroll
    for (int ii = 0; ii < 3; ++ii)
#pragma unroll
      for (int kk = 0; kk < 3; ++kk)
        Rn[ii * 3 + kk] = R[ii * 3 + 0] * Ru[kk] + R[ii * 3 + 1] * Ru[3 + kk] + R[ii * 3 + 2] * Ru[6 + kk];
#pragma unroll
    for (int e = 0; e < 9; ++e) rots[r * 9 + e] = Rn[e];
    trans[r * 3 + 0] = tn0; trans[r * 3 + 1] = tn1; trans[r * 3 + 2] = tn2;
  }
}

__global__ void relu_h_k(const float* __restrict__ x, h16* __restrict__ y, int n) {
  int i = blockIdx.x * 256 + threadIdx.x;
  if (i < n) y[i] = (h16)fmaxf(x[i], 0.f);
}

__global__ void tors_k(const float* __restrict__ un, float* __restrict__ out) {
  int i = blockIdx.x, t = threadIdx.x;
  if (t < 14) {
    int k = t >> 1;
    float u0 = un[i * 14 + 2 * k], u1 = un[i * 14 + 2 * k + 1];
    float nrm = sqrtf(fmaxf(u0 * u0 + u1 * u1, 1e-20f));
    out[i * 14 + t] = un[i * 14 + t] / nrm;
  }
}

__global__ void copy_f32_k(const float* __restrict__ src, float* __restrict__ dst, int n) {
  int i = blockIdx.x * 256 + threadIdx.x;
  if (i < n) dst[i] = src[i];
}

// ----------------------------- host ----------------------------------------
// Input order assumption: s, z, s_mask, aatype, then params flattened as a JAX
// pytree (dict keys sorted alphabetically at each level, lists in order).

extern "C" void kernel_launch(void* const* d_in, const int* in_sizes, int n_in,
                              void* d_out, int out_size, void* d_ws, size_t ws_size,
                              hipStream_t stream) {
  (void)in_sizes; (void)n_in; (void)out_size; (void)ws_size;
  const float* s_in  = (const float*)d_in[0];
  const float* z_in  = (const float*)d_in[1];
  const float* maskp = (const float*)d_in[2];
  // d_in[3] = aatype (unused by reference)
  const float* Wai   = (const float*)d_in[4];
  const float* Wain  = (const float*)d_in[5];
  const float* Wao   = (const float*)d_in[6];
  const float* Wbb   = (const float*)d_in[7];
  const float* Win   = (const float*)d_in[8];
  const float* bai   = (const float*)d_in[9];
  const float* bain  = (const float*)d_in[10];
  const float* bao   = (const float*)d_in[11];
  const float* bbb   = (const float*)d_in[12];
  const float* bin   = (const float*)d_in[13];
  const float* iWb   = (const float*)d_in[14];
  const float* iWkv  = (const float*)d_in[15];
  const float* iWkv3 = (const float*)d_in[16];
  const float* iWo   = (const float*)d_in[17];
  const float* iWq   = (const float*)d_in[18];
  const float* iWq3  = (const float*)d_in[19];
  const float* ibb   = (const float*)d_in[20];
  const float* ibkv  = (const float*)d_in[21];
  const float* ibkv3 = (const float*)d_in[22];
  const float* ibo   = (const float*)d_in[23];
  const float* ibq   = (const float*)d_in[24];
  const float* ibq3  = (const float*)d_in[25];
  const float* ihw   = (const float*)d_in[26];
  const float* ln_i_b = (const float*)d_in[27];
  const float* ln_i_g = (const float*)d_in[28];
  const float* ln_s_b = (const float*)d_in[29];
  const float* ln_s_g = (const float*)d_in[30];
  const float* ln_t_b = (const float*)d_in[31];
  const float* ln_t_g = (const float*)d_in[32];
  const float* ln_z_b = (const float*)d_in[33];
  const float* ln_z_g = (const float*)d_in[34];
  const float* rW1[2] = {(const float*)d_in[35], (const float*)d_in[39]};
  const float* rW2[2] = {(const float*)d_in[36], (const float*)d_in[40]};
  const float* rb1[2] = {(const float*)d_in[37], (const float*)d_in[41]};
  const float* rb2[2] = {(const float*)d_in[38], (const float*)d_in[42]};
  const float* tW1 = (const float*)d_in[43];
  const float* tW2 = (const float*)d_in[44];
  const float* tW3 = (const float*)d_in[45];
  const float* tb1 = (const float*)d_in[46];
  const float* tb2 = (const float*)d_in[47];
  const float* tb3 = (const float*)d_in[48];
  float* out = (float*)d_out;

  char* base = (char*)d_ws;
  size_t cur = 0;
  auto alloc = [&](size_t bytes) -> void* {
    void* p = base + cur;
    cur += (bytes + 255) & ~(size_t)255;
    return p;
  };

  h16*   zlT    = (h16*)  alloc((size_t)512 * 128 * 512 * 2);  // LN(z)^T f16, L2-resident
  h16*   Bterm  = (h16*)  alloc((size_t)12 * 512 * 512 * 2);
  h16*   P      = (h16*)  alloc((size_t)512 * 12 * 512 * 2);
  h16*   AttA   = (h16*)  alloc((size_t)512 * 12 * 32 * 2);
  h16*   AttB   = (h16*)  alloc((size_t)512 * 12 * 32 * 2);
  float* rowq   = (float*)alloc((size_t)512 * 12 * 4);
  float* rowk   = (float*)alloc((size_t)512 * 12 * 4);
  h16*   VV3t   = (h16*)  alloc((size_t)12 * 40 * 512 * 2);
  h16*   Of     = (h16*)  alloc((size_t)512 * 2112 * 2);
  float* O3raw  = (float*)alloc((size_t)512 * 288 * 4);
  float* s_init = (float*)alloc((size_t)512 * 384 * 4);
  h16*   sinit16= (h16*)  alloc((size_t)512 * 384 * 2);
  float* s_cur  = (float*)alloc((size_t)512 * 384 * 4);
  h16*   s16    = (h16*)  alloc((size_t)512 * 384 * 2);
  float* qraw   = (float*)alloc((size_t)512 * 192 * 4);
  float* kvraw  = (float*)alloc((size_t)512 * 384 * 4);
  float* q3raw  = (float*)alloc((size_t)512 * 144 * 4);
  float* kv3raw = (float*)alloc((size_t)512 * 432 * 4);
  h16*   t1h    = (h16*)  alloc((size_t)512 * 384 * 2);
  h16*   t2h    = (h16*)  alloc((size_t)512 * 384 * 2);
  float* rotsb  = (float*)alloc((size_t)512 * 9 * 4);
  float* transb = (float*)alloc((size_t)512 * 3 * 4);
  float* a32    = (float*)alloc((size_t)512 * 128 * 4);
  h16*   ra16   = (h16*)  alloc((size_t)512 * 128 * 2);
  h16*   th16   = (h16*)  alloc((size_t)512 * 128 * 2);
  float* un     = (float*)alloc((size_t)512 * 14 * 4);
  h16*   rs16   = (h16*)  alloc((size_t)512 * 384 * 2);
  h16*   rsi16  = (h16*)  alloc((size_t)512 * 384 * 2);
  h16* WinT  = (h16*)alloc((size_t)384 * 384 * 2);
  h16* WqT   = (h16*)alloc((size_t)384 * 192 * 2);
  h16* WkvT  = (h16*)alloc((size_t)384 * 384 * 2);
  h16* Wq3T  = (h16*)alloc((size_t)384 * 144 * 2);
  h16* Wkv3T = (h16*)alloc((size_t)384 * 432 * 2);
  h16* WoT   = (h16*)alloc((size_t)2112 * 384 * 2);
  h16* tW1T  = (h16*)alloc((size_t)384 * 384 * 2);
  h16* tW2T  = (h16*)alloc((size_t)384 * 384 * 2);
  h16* tW3T  = (h16*)alloc((size_t)384 * 384 * 2);
  h16* WainT = (h16*)alloc((size_t)384 * 128 * 2);
  h16* WaiT  = (h16*)alloc((size_t)384 * 128 * 2);
  h16* rW1T[2] = {(h16*)alloc((size_t)128 * 128 * 2), (h16*)alloc((size_t)128 * 128 * 2)};
  h16* rW2T[2] = {(h16*)alloc((size_t)128 * 128 * 2), (h16*)alloc((size_t)128 * 128 * 2)};
  h16* WaoT  = (h16*)alloc((size_t)128 * 14 * 2);

  auto wtr = [&](const float* W, h16* Wt, int din, int dout) {
    int n = din * dout;
    wtr_k<<<(n + 255) / 256, 256, 0, stream>>>(W, Wt, din, dout);
  };
  auto gemm = [&](const h16* A, int lda, const h16* Bt, int ldb, const float* bias,
                  const float* Cin, int ldcin, float* Cf, h16* Ch, int ldc,
                  int M, int K, int N, int relu) {
    int tiles = (M / 16) * ((N + 31) / 32);
    gemm16_k<<<tiles, 32, 0, stream>>>(A, lda, Bt, ldb, bias, Cin, ldcin, Cf, Ch, ldc, M, K, N, relu);
  };

  // Weight convert+transpose (f32 [din][dout] -> f16 [dout][din]) every call.
  wtr(Win, WinT, 384, 384);   wtr(iWq, WqT, 384, 192);    wtr(iWkv, WkvT, 384, 384);
  wtr(iWq3, Wq3T, 384, 144);  wtr(iWkv3, Wkv3T, 384, 432); wtr(iWo, WoT, 2112, 384);
  wtr(tW1, tW1T, 384, 384);   wtr(tW2, tW2T, 384, 384);   wtr(tW3, tW3T, 384, 384);
  wtr(Wain, WainT, 384, 128); wtr(Wai, WaiT, 384, 128);
  wtr(rW1[0], rW1T[0], 128, 128); wtr(rW2[0], rW2T[0], 128, 128);
  wtr(rW1[1], rW1T[1], 128, 128); wtr(rW2[1], rW2T[1], 128, 128);
  wtr(Wao, WaoT, 128, 14);

  // LN(s) -> s_init (kept for the final angle-resnet branch) + f16 for Win GEMM.
  ln_row_k<<<512, 32, 0, stream>>>(s_in, ln_s_g, ln_s_b, s_init, sinit16, 384);
  // LN(z) once: f16 transposed copy + fused B = zl@Wb+bb.
  ln_z_k<<<512 * 512, 32, 0, stream>>>(z_in, ln_z_g, ln_z_b, iWb, ibb, zlT, Bterm);
  // s = s_init @ Win + bin
  gemm(sinit16, 384, WinT, 384, bin, nullptr, 0, s_cur, s16, 384, 512, 384, 384, 0);
  frames_init_k<<<8, 64, 0, stream>>>(rotsb, transb);

  for (int blk = 0; blk < 4; ++blk) {
    // QKV / point projections
    gemm(s16, 384, WqT, 384, ibq, nullptr, 0, qraw, nullptr, 192, 512, 384, 192, 0);
    gemm(s16, 384, WkvT, 384, ibkv, nullptr, 0, kvraw, nullptr, 384, 512, 384, 384, 0);
    gemm(s16, 384, Wq3T, 384, ibq3, nullptr, 0, q3raw, nullptr, 144, 512, 384, 144, 0);
    gemm(s16, 384, Wkv3T, 384, ibkv3, nullptr, 0, kv3raw, nullptr, 432, 512, 384, 432, 0);
    qkv_prep_k<<<512, 448, 0, stream>>>(qraw, kvraw, q3raw, kv3raw, ihw, rotsb, transb,
                                        AttA, AttB, rowq, rowk, VV3t);
    attn_k<<<dim3(32, 12), 32, 0, stream>>>(AttA, AttB, rowq, rowk, Bterm, maskp, P);
    oo3_k<<<dim3(32, 12), 32, 0, stream>>>(P, VV3t, Of, O3raw);
    op_k<<<dim3(512, 8), 32, 0, stream>>>(P, zlT, Of);
    o3fin_k<<<512, 96, 0, stream>>>(O3raw, rotsb, transb, Of);
    // s = s + Of@Wo+bo ; LN(ln_i)
    gemm(Of, 2112, WoT, 2112, ibo, s_cur, 384, s_cur, nullptr, 384, 512, 2112, 384, 0);
    ln_row_k<<<512, 32, 0, stream>>>(s_cur, ln_i_g, ln_i_b, s_cur, s16, 384);
    // transition (relu, relu, linear) + residual ; LN(ln_t)
    gemm(s16, 384, tW1T, 384, tb1, nullptr, 0, nullptr, t1h, 384, 512, 384, 384, 1);
    gemm(t1h, 384, tW2T, 384, tb2, nullptr, 0, nullptr, t2h, 384, 512, 384, 384, 1);
    gemm(t2h, 384, tW3T, 384, tb3, s_cur, 384, s_cur, nullptr, 384, 512, 384, 384, 0);
    ln_row_k<<<512, 32, 0, stream>>>(s_cur, ln_t_g, ln_t_b, s_cur, s16, 384);
    // backbone frame update
    bbq_k<<<512, 32, 0, stream>>>(s_cur, Wbb, bbb, rotsb, transb);
  }

  // Angle resnet
  relu_h_k<<<(196608 + 255) / 256, 256, 0, stream>>>(s_cur, rs16, 196608);
  relu_h_k<<<(196608 + 255) / 256, 256, 0, stream>>>(s_init, rsi16, 196608);
  gemm(rs16, 384, WainT, 384, bain, nullptr, 0, a32, nullptr, 128, 512, 384, 128, 0);
  gemm(rsi16, 384, WaiT, 384, bai, a32, 128, a32, nullptr, 128, 512, 384, 128, 0);
  for (int l = 0; l < 2; ++l) {
    relu_h_k<<<(65536 + 255) / 256, 256, 0, stream>>>(a32, ra16, 65536);
    gemm(ra16, 128, rW1T[l], 128, rb1[l], nullptr, 0, nullptr, th16, 128, 512, 128, 128, 1);
    gemm(th16, 128, rW2T[l], 128, rb2[l], a32, 128, a32, nullptr, 128, 512, 128, 128, 0);
  }
  relu_h_k<<<(65536 + 255) / 256, 256, 0, stream>>>(a32, ra16, 65536);
  gemm(ra16, 128, WaoT, 128, bao, nullptr, 0, un, nullptr, 14, 512, 128, 14, 0);

  // Outputs: s (512x384) | rots (512x3x3) | trans (512x3) | tors (512x7x2)
  copy_f32_k<<<(196608 + 255) / 256, 256, 0, stream>>>(s_cur, out, 196608);
  copy_f32_k<<<(4608 + 255) / 256, 256, 0, stream>>>(rotsb, out + 196608, 4608);
  copy_f32_k<<<(1536 + 255) / 256, 256, 0, stream>>>(transb, out + 201216, 1536);
  tors_k<<<512, 16, 0, stream>>>(un, out + 202752);
}
